// DynamicSemanticHead_35966056136950
// MI455X (gfx1250) — matrix operand order repaired and checked by reading, not compiled, not run on hardware
//
#include <hip/hip_runtime.h>
#include <hip/hip_bf16.h>
#include <stdint.h>

typedef __attribute__((ext_vector_type(16))) _Float16 v16h;
typedef __attribute__((ext_vector_type(8)))  float    v8f;

#define B_    2
#define CIN_  16
#define COUT_ 8
#define D_    96
#define TDIM_ 256
#define E_    8
#define NTAP  27
#define NFRAG 14                          // ceil(27 taps / 2) K=32 WMMA steps
#define NOUT_MAIN (B_*COUT_*D_*D_*D_)     // 14,155,776

// spatial tiling for the conv kernel
#define TY 4
#define TX 16
#define PLY (TY+2)                        // 6
#define PLX (TX+2)                        // 18
#define PLANE_H (PLY*PLX*CIN_)            // 1728 halves per z-plane

// workspace layout (floats):
//  [0,16)  cb[b][co]          (mixed bias)
//  [16,32) g2[b][co] = 1+gamma
//  byte 128+: f16 A-fragments, [b][frag(14)][lane(32)][16 halves]
#define AFRAG_HALVES (NFRAG*32*16)        // 7168 per sample

// ---------------------------------------------------------------------------
// Kernel 1: router + modulator MLPs, softmax, expert mixing, and pre-swizzle
// of the mixed conv weights into WMMA 16x32 f16 A-fragment lane layout.
// ---------------------------------------------------------------------------
__global__ __launch_bounds__(256)
void k1_route_mix(const float* __restrict__ text,
                  const float* __restrict__ wbasis,   // [E,COUT,CIN,27]
                  const float* __restrict__ bbasis,   // [E,COUT]
                  const float* __restrict__ rw1, const float* __restrict__ rb1,
                  const float* __restrict__ rw2, const float* __restrict__ rb2,
                  const float* __restrict__ mw1, const float* __restrict__ mb1,
                  const float* __restrict__ mw2, const float* __restrict__ mb2,
                  float* __restrict__ out,            // full d_out (tail used)
                  float* __restrict__ wsf)
{
    __shared__ float sh_h[B_][64];
    __shared__ float sh_g[B_][64];
    __shared__ float sh_logit[B_][E_];
    __shared__ float sh_rw[B_][E_];
    __shared__ float sh_cw[B_*COUT_*CIN_*NTAP];      // 6912 f32 mixed weights
    const int tid = threadIdx.x;

    // hidden layers: 128 router dots + 128 modulator dots, length 256
    if (tid < 128) {
        int b = tid >> 6, j = tid & 63;
        float s = rb1[j];
        const float* t = text + b*TDIM_;
        for (int k = 0; k < TDIM_; ++k) s += t[k]*rw1[k*64+j];
        sh_h[b][j] = s > 0.f ? s : 0.1f*s;           // leaky_relu 0.1
    } else {
        int t2 = tid - 128; int b = t2 >> 6, j = t2 & 63;
        float s = mb1[j];
        const float* t = text + b*TDIM_;
        for (int k = 0; k < TDIM_; ++k) s += t[k]*mw1[k*64+j];
        sh_g[b][j] = s > 0.f ? s : 0.1f*s;
    }
    __syncthreads();

    if (tid < 16) {                                  // router logits
        int b = tid >> 3, e = tid & 7;
        float s = rb2[e];
        for (int k = 0; k < 64; ++k) s += sh_h[b][k]*rw2[k*E_+e];
        sh_logit[b][e] = s;
        out[NOUT_MAIN + b*E_ + e] = s;               // router_logits output
    } else if (tid >= 32 && tid < 48) {              // channel gates
        int t2 = tid - 32; int b = t2 >> 3, co = t2 & 7;
        float s = mb2[co];
        for (int k = 0; k < 64; ++k) s += sh_g[b][k]*mw2[k*COUT_+co];
        float gamma = 1.f/(1.f + __expf(-s));
        wsf[16 + b*COUT_ + co] = 1.f + gamma;
    }
    __syncthreads();

    if (tid < B_) {                                  // softmax over experts
        int b = tid;
        float m = sh_logit[b][0];
        for (int e = 1; e < E_; ++e) m = fmaxf(m, sh_logit[b][e]);
        float ex[E_], sum = 0.f;
        for (int e = 0; e < E_; ++e) { ex[e] = __expf(sh_logit[b][e]-m); sum += ex[e]; }
        for (int e = 0; e < E_; ++e) {
            float r = ex[e]/sum;
            sh_rw[b][e] = r;
            out[NOUT_MAIN + B_*E_ + b*E_ + e] = r;   // routing_weights output
        }
    }
    __syncthreads();

    // mixed conv weights cw[b][co][ci][tap]
    for (int idx = tid; idx < B_*COUT_*CIN_*NTAP; idx += 256) {
        int b  = idx / (COUT_*CIN_*NTAP);
        int r  = idx % (COUT_*CIN_*NTAP);
        int co = r / (CIN_*NTAP);
        int r2 = r % (CIN_*NTAP);
        int ci = r2 / NTAP;
        int t  = r2 % NTAP;
        float s = 0.f;
        for (int e = 0; e < E_; ++e)
            s += sh_rw[b][e] * wbasis[((e*COUT_+co)*CIN_+ci)*NTAP + t];
        sh_cw[idx] = s;
    }
    if (tid < 16) {                                  // mixed bias
        int b = tid >> 3, co = tid & 7;
        float s = 0.f;
        for (int e = 0; e < E_; ++e) s += sh_rw[b][e]*bbasis[e*COUT_+co];
        wsf[b*COUT_ + co] = s;
    }
    __syncthreads();

    // Swizzle into WMMA f16 A-fragment layout (16-bit A 16x32, contiguous K order, matched by B):
    // lane = hh*16+m; element j -> K = hh*16 + j;
    // tap = 2*frag + (K>>4), ci = K&15, row M = m (zero for M>=8 / tap 27).
    _Float16* af = (_Float16*)(wsf + 32);
    for (int idx = tid; idx < B_*AFRAG_HALVES; idx += 256) {
        int b    = idx / AFRAG_HALVES;
        int r    = idx % AFRAG_HALVES;
        int f    = r / 512;
        int r2   = r % 512;
        int lane = r2 >> 4;
        int j    = r2 & 15;
        int hh   = lane >> 4, m = lane & 15;
        int K    = hh*16 + j;
        int tap  = 2*f + (K >> 4);
        int ci   = K & 15;
        float v  = 0.f;
        if (m < COUT_ && tap < NTAP)
            v = sh_cw[((b*COUT_+m)*CIN_+ci)*NTAP + tap];
        af[idx] = (_Float16)v;
    }
}

// ---------------------------------------------------------------------------
// Kernel 2: dynamic 3x3x3 conv as WMMA f32<-f16 GEMM (M=16 pad, N=16 px,
// K=432 as 14 tap-pair steps of K=32), rolling 3-plane f16 LDS tile.
// ---------------------------------------------------------------------------
__global__ __launch_bounds__(128)
void k2_dynconv(const float* __restrict__ x,
                float* __restrict__ out,
                const float* __restrict__ wsf)
{
    __shared__ _Float16 lds_w[AFRAG_HALVES] __attribute__((aligned(32)));   // 14 KB
    __shared__ _Float16 lds_p[3*PLANE_H]    __attribute__((aligned(32)));   // 10.1 KB
    __shared__ _Float16 lds_zero[16]        __attribute__((aligned(32)));

    const int tid  = threadIdx.x;
    const int x0   = blockIdx.x * TX;
    const int y0   = blockIdx.y * TY;
    const int b    = blockIdx.z;
    const int lane = tid & 31;
    const int wv   = tid >> 5;        // wave id = local y row (0..3)
    const int hh   = lane >> 4;       // lane half (selects tap of the pair)
    const int n    = lane & 15;       // output pixel within tile (N dim)

    if (tid < 16) lds_zero[tid] = (_Float16)0.f;

    // per-sample A-fragments -> LDS (uint32 copies, served from L2)
    {
        const uint32_t* src = (const uint32_t*)(wsf + 32) + b*(AFRAG_HALVES/2);
        uint32_t* dst = (uint32_t*)lds_w;
        for (int i = tid; i < AFRAG_HALVES/2; i += 128) dst[i] = src[i];
    }

    float cbv[8], g2v[8];
#pragma unroll
    for (int r = 0; r < 8; ++r) { cbv[r] = wsf[b*8+r]; g2v[r] = wsf[16+b*8+r]; }

    const float* xb = x + (size_t)b*CIN_*D_*D_*D_;

    auto load_plane = [&](int zp) {
        int slot = (zp + 3) % 3;
        _Float16* dst = lds_p + slot*PLANE_H;
        for (int i = tid; i < PLANE_H; i += 128) {
            int ci = i & 15;
            int xx = (i >> 4) % PLX;
            int yy = i / (16*PLX);
            int gy = y0 - 1 + yy;
            int gx = x0 - 1 + xx;
            float v = 0.f;
            if ((unsigned)zp < D_ && (unsigned)gy < D_ && (unsigned)gx < D_)
                v = xb[((size_t)ci*D_ + zp)*D_*D_ + (size_t)gy*D_ + gx];
            dst[i] = (_Float16)v;    // [yy][xx][ci] -> 32B-contiguous channels
        }
    };

    load_plane(-1);      // zero halo plane
    load_plane(0);

    for (int z = 0; z < D_; ++z) {
        load_plane(z + 1);
        if (z + 2 < D_) {            // pull z+2 plane toward L2/L0
            __builtin_prefetch(xb + (size_t)(tid & 15)*D_*D_*D_
                                  + (size_t)(z + 2)*D_*D_
                                  + (size_t)y0*D_ + x0, 0, 1);
        }
        __syncthreads();

        v8f acc = {};
#pragma unroll
        for (int f = 0; f < NFRAG; ++f) {
            v16h a = *(const v16h*)(lds_w + (f*32 + lane)*16);
            int t = 2*f + hh;                       // this lane-half's tap
            const _Float16* bp = lds_zero;          // pad tap 27 (keep B finite)
            if (t < NTAP) {
                int kz = t/9, rr = t%9, ky = rr/3, kx = rr%3;
                int slot = (z + kz - 1 + 3) % 3;
                bp = lds_p + ((slot*PLY + (wv + ky))*PLX + (n + kx))*CIN_;
            }
            v16h bv = *(const v16h*)bp;             // 16 channels, 32B ds read
            acc = __builtin_amdgcn_wmma_f32_16x16x32_f16(
                      false, a, false, bv, (short)0, acc, false, false);
        }

        if (lane < 16) {                            // valid D rows: M=0..7
            union { v8f v; float f[8]; } u; u.v = acc;
            int gy = y0 + wv;
#pragma unroll
            for (int r = 0; r < 8; ++r) {
                size_t oi = ((((size_t)(b*COUT_ + r))*D_ + z)*D_ + gy)*D_ + (x0 + n);
                out[oi] = (u.f[r] + cbv[r]) * g2v[r];
            }
        }
        __syncthreads();                            // protect plane reuse
    }
}

extern "C" void kernel_launch(void* const* d_in, const int* in_sizes, int n_in,
                              void* d_out, int out_size, void* d_ws, size_t ws_size,
                              hipStream_t stream) {
    const float* x    = (const float*)d_in[0];
    const float* text = (const float*)d_in[1];
    const float* wb   = (const float*)d_in[2];
    const float* bb   = (const float*)d_in[3];
    const float* rw1  = (const float*)d_in[4];
    const float* rb1  = (const float*)d_in[5];
    const float* rw2  = (const float*)d_in[6];
    const float* rb2  = (const float*)d_in[7];
    const float* mw1  = (const float*)d_in[8];
    const float* mb1  = (const float*)d_in[9];
    const float* mw2  = (const float*)d_in[10];
    const float* mb2  = (const float*)d_in[11];
    float* outp = (float*)d_out;
    float* wsf  = (float*)d_ws;

    k1_route_mix<<<1, 256, 0, stream>>>(text, wb, bb, rw1, rb1, rw2, rb2,
                                        mw1, mb1, mw2, mb2, outp, wsf);
    k2_dynconv<<<dim3(D_/TX, D_/TY, B_), 128, 0, stream>>>(x, outp, wsf);
}
